// AttentionBlock_12541304504862
// MI455X (gfx1250) — matrix operand order repaired and checked by reading, not compiled, not run on hardware
//
#include <hip/hip_runtime.h>

typedef __attribute__((ext_vector_type(16))) _Float16 v16h;
typedef __attribute__((ext_vector_type(8)))  float    v8f;

#define Bb   8
#define Cc   64
#define Nn   4096
#define Gg   8
#define CPG  8
#define EPSV 1e-5f

// ---------- helpers ----------
union V16HU { float4 f4[2]; v16h v; };
union H8U   { float4 f4;    _Float16 h[8]; };

__device__ __forceinline__ float4 ld16g(const _Float16* p) { return *(const float4*)p; }

__device__ __forceinline__ v16h mkv(float4 a, float4 b) {
    V16HU u; u.f4[0] = a; u.f4[1] = b; return u.v;
}

__device__ __forceinline__ v8f wmma16(v16h a, v16h b, v8f c) {
    // D = A(16x32 f16) * B(32x16 f16) + C(16x16 f32)
    return __builtin_amdgcn_wmma_f32_16x16x32_f16(false, a, false, b, (short)0, c, false, false);
}

__device__ __forceinline__ float rmax16(float v) {
#pragma unroll
    for (int m = 8; m > 0; m >>= 1) v = fmaxf(v, __shfl_xor(v, m, 16));
    return v;
}
__device__ __forceinline__ float rsum16(float v) {
#pragma unroll
    for (int m = 8; m > 0; m >>= 1) v += __shfl_xor(v, m, 16);
    return v;
}

// ---------- 0: convert weights to f16 ----------
__global__ void cvtw_kernel(const float* __restrict__ Wq, const float* __restrict__ Wk,
                            const float* __restrict__ Wv, const float* __restrict__ Wp,
                            _Float16* __restrict__ w16) {
    int i = blockIdx.x * 256 + threadIdx.x;            // 0..4095
    w16[i]          = (_Float16)Wq[i];
    w16[4096 + i]   = (_Float16)Wk[i];
    w16[8192 + i]   = (_Float16)Wv[i];
    w16[12288 + i]  = (_Float16)Wp[i];
}

// ---------- 1: GroupNorm -> h f16 [B][N][C] (pixel-major) ----------
__global__ void gn_kernel(const float* __restrict__ x, const float* __restrict__ gw,
                          const float* __restrict__ gb, _Float16* __restrict__ h16) {
    int bg = blockIdx.x; int b = bg >> 3; int g = bg & 7;
    const float* xb = x + ((size_t)(b * Cc + g * CPG)) * Nn;
    __shared__ float s1[256], s2[256];
    float sum = 0.f, sq = 0.f;
#pragma unroll
    for (int c = 0; c < CPG; c++) {
        const float* xc = xb + (size_t)c * Nn;
        for (int n = threadIdx.x; n < Nn; n += 256) { float v = xc[n]; sum += v; sq += v * v; }
    }
    s1[threadIdx.x] = sum; s2[threadIdx.x] = sq; __syncthreads();
    for (int s = 128; s > 0; s >>= 1) {
        if (threadIdx.x < (unsigned)s) { s1[threadIdx.x] += s1[threadIdx.x + s]; s2[threadIdx.x] += s2[threadIdx.x + s]; }
        __syncthreads();
    }
    const float inv_cnt = 1.0f / (float)(CPG * Nn);
    float mean = s1[0] * inv_cnt;
    float var  = s2[0] * inv_cnt - mean * mean;
    float rstd = rsqrtf(var + EPSV);
    float wv[CPG], bv[CPG];
#pragma unroll
    for (int c = 0; c < CPG; c++) {
        float w = gw[g * CPG + c] * rstd;
        wv[c] = w; bv[c] = gb[g * CPG + c] - mean * w;
    }
    for (int n = threadIdx.x; n < Nn; n += 256) {
        H8U u;
#pragma unroll
        for (int c = 0; c < CPG; c++) u.h[c] = (_Float16)(xb[(size_t)c * Nn + n] * wv[c] + bv[c]);
        *(float4*)(h16 + ((size_t)(b * Nn + n) * Cc + g * CPG)) = u.f4;   // 16B store
    }
}

// ---------- 2: QKV 1x1 convs (WMMA) ----------
// q,k -> f16 [B][N][64] row-major (q pre-scaled by 1/sqrt(C)); v -> f16 [B][64][N]
__global__ __launch_bounds__(32) void qkv_kernel(const _Float16* __restrict__ h16,
                                                 const _Float16* __restrict__ w16,
                                                 const float* __restrict__ bq,
                                                 const float* __restrict__ bk,
                                                 const float* __restrict__ bv,
                                                 _Float16* __restrict__ q16,
                                                 _Float16* __restrict__ k16,
                                                 _Float16* __restrict__ vT16) {
    int blk = blockIdx.x; int b = blk >> 8; int n0 = (blk & 255) * 16;
    int lane = threadIdx.x & 31; int ln = lane & 15; int hi = lane >> 4;

    // B-fragments of h (shared by q,k,v): lane = pixel, elements = channels
    const _Float16* hrow = h16 + (size_t)(b * Nn + n0 + ln) * Cc;
    v16h hb0 = mkv(ld16g(hrow + 16 * hi),      ld16g(hrow + 16 * hi + 8));       // c 0..31
    v16h hb1 = mkv(ld16g(hrow + 32 + 16 * hi), ld16g(hrow + 40 + 16 * hi));      // c 32..63

    const _Float16* Ws[3] = { w16, w16 + 4096, w16 + 8192 };
    const float*    Bs[3] = { bq, bk, bv };
    const float     Sc[3] = { 0.125f, 1.0f, 1.0f };   // 1/sqrt(64) folded into q

    for (int mat = 0; mat < 3; mat++) {
        const _Float16* W = Ws[mat]; const float* bias = Bs[mat]; float sc = Sc[mat];
#pragma unroll
        for (int mt = 0; mt < 4; mt++) {
            const _Float16* wrow = W + (size_t)(mt * 16 + ln) * Cc;
            v16h wa0 = mkv(ld16g(wrow + 16 * hi),      ld16g(wrow + 16 * hi + 8));
            v16h wa1 = mkv(ld16g(wrow + 32 + 16 * hi), ld16g(wrow + 40 + 16 * hi));
            v8f z = {};
            v8f d = wmma16(wa0, hb0, wmma16(wa1, hb1, z));
            if (mat < 2) {
                // row-major store: lane -> pixel n0+ln, channels mt*16+8*hi+[0..8)
                H8U u;
#pragma unroll
                for (int j = 0; j < 8; j++)
                    u.h[j] = (_Float16)((d[j] + bias[mt * 16 + 8 * hi + j]) * sc);
                _Float16* dst = (mat == 0 ? q16 : k16) +
                                (size_t)(b * Nn + n0 + ln) * Cc + mt * 16 + 8 * hi;
                *(float4*)dst = u.f4;
            } else {
                // transposed store: vT[b][c][n]
#pragma unroll
                for (int j = 0; j < 8; j++) {
                    int c = mt * 16 + 8 * hi + j;
                    vT16[(size_t)(b * Cc + c) * Nn + n0 + ln] = (_Float16)(d[j] + bias[c]);
                }
            }
        }
    }
}

// ---------- 3: flash attention (1 wave = 16 query rows) ----------
__global__ __launch_bounds__(32) void attn_kernel(const _Float16* __restrict__ q16,
                                                  const _Float16* __restrict__ k16,
                                                  const _Float16* __restrict__ vT16,
                                                  _Float16* __restrict__ O16) {
    int blk = blockIdx.x; int b = blk >> 8; int q0 = (blk & 255) * 16;
    int lane = threadIdx.x & 31; int ln = lane & 15; int hi = lane >> 4;
    __shared__ _Float16 pbuf[16 * 32];

    const _Float16* qrow = q16 + (size_t)(b * Nn + q0 + ln) * Cc;
    v16h qa0 = mkv(ld16g(qrow + 16 * hi),      ld16g(qrow + 16 * hi + 8));   // d 0..31
    v16h qa1 = mkv(ld16g(qrow + 32 + 16 * hi), ld16g(qrow + 40 + 16 * hi));   // d 32..63

    v8f o0 = {}, o1 = {}, o2 = {}, o3 = {};
    float mrow[8], lrow[8];
#pragma unroll
    for (int j = 0; j < 8; j++) { mrow[j] = -1e30f; lrow[j] = 0.f; }

    const _Float16* kbase = k16 + (size_t)b * Nn * Cc;
    const _Float16* vbase = vT16 + (size_t)b * Cc * Nn;

    for (int kt = 0; kt < Nn / 32; kt++) {
        int key0 = kt * 32;
        // ---- S = Q * K^T for two 16-key subtiles ----
        const _Float16* kr0 = kbase + (size_t)(key0 + ln) * Cc;
        const _Float16* kr1 = kbase + (size_t)(key0 + 16 + ln) * Cc;
        __builtin_prefetch(kr0 + 32 * Cc, 0, 1);    // next K tile -> global_prefetch
        v8f z = {};
        v16h kb;
        kb = mkv(ld16g(kr0 + 32 + 16 * hi), ld16g(kr0 + 40 + 16 * hi));
        v8f s0 = wmma16(qa1, kb, z);
        kb = mkv(ld16g(kr0 + 16 * hi),      ld16g(kr0 + 16 * hi + 8));
        s0 = wmma16(qa0, kb, s0);
        kb = mkv(ld16g(kr1 + 32 + 16 * hi), ld16g(kr1 + 40 + 16 * hi));
        v8f s1 = wmma16(qa1, kb, z);
        kb = mkv(ld16g(kr1 + 16 * hi),      ld16g(kr1 + 16 * hi + 8));
        s1 = wmma16(qa0, kb, s1);

        // ---- online softmax (row = j + 8*hi; single width-16 reduction per row) ----
#pragma unroll
        for (int j = 0; j < 8; j++) {
            float mn = fmaxf(mrow[j], rmax16(fmaxf(s0[j], s1[j])));
            float alpha = __expf(mrow[j] - mn);
            mrow[j] = mn;
            float p0 = __expf(s0[j] - mn);
            float p1 = __expf(s1[j] - mn);
            lrow[j] = lrow[j] * alpha + rsum16(p0 + p1);
            o0[j] *= alpha; o1[j] *= alpha; o2[j] *= alpha; o3[j] *= alpha;
            pbuf[(j + 8 * hi) * 32 + ln]      = (_Float16)p0;
            pbuf[(j + 8 * hi) * 32 + 16 + ln] = (_Float16)p1;
        }
        __syncthreads();
        // ---- P (16x32) as A-fragment from LDS ----
        const _Float16* prow = pbuf + ln * 32;
        v16h pa = mkv(ld16g(prow + 16 * hi), ld16g(prow + 16 * hi + 8));
        __syncthreads();
        // ---- O += P * V ----
#pragma unroll
        for (int ct = 0; ct < 4; ct++) {
            const _Float16* vrow = vbase + (size_t)(ct * 16 + ln) * Nn + key0 + 16 * hi;
            v16h vb = mkv(ld16g(vrow), ld16g(vrow + 8));
            if (ct == 0) o0 = wmma16(pa, vb, o0);
            else if (ct == 1) o1 = wmma16(pa, vb, o1);
            else if (ct == 2) o2 = wmma16(pa, vb, o2);
            else o3 = wmma16(pa, vb, o3);
        }
    }

    // ---- normalize and store O f16 [B][N][64] ----
    float inv[8];
#pragma unroll
    for (int j = 0; j < 8; j++) inv[j] = 1.0f / lrow[j];
#pragma unroll
    for (int j = 0; j < 8; j++) {
        size_t base = (size_t)(b * Nn + q0 + j + 8 * hi) * Cc + ln;
        O16[base]      = (_Float16)(o0[j] * inv[j]);
        O16[base + 16] = (_Float16)(o1[j] * inv[j]);
        O16[base + 32] = (_Float16)(o2[j] * inv[j]);
        O16[base + 48] = (_Float16)(o3[j] * inv[j]);
    }
}

// ---------- 4: output projection + bias + residual ----------
__global__ __launch_bounds__(32) void proj_kernel(const _Float16* __restrict__ O16,
                                                  const _Float16* __restrict__ wp16,
                                                  const float* __restrict__ bp,
                                                  const float* __restrict__ x,
                                                  float* __restrict__ out) {
    int blk = blockIdx.x; int b = blk >> 8; int n0 = (blk & 255) * 16;
    int lane = threadIdx.x & 31; int ln = lane & 15; int hi = lane >> 4;

    const _Float16* orow = O16 + (size_t)(b * Nn + n0 + ln) * Cc;
    v16h ob0 = mkv(ld16g(orow + 16 * hi),      ld16g(orow + 16 * hi + 8));
    v16h ob1 = mkv(ld16g(orow + 32 + 16 * hi), ld16g(orow + 40 + 16 * hi));

#pragma unroll
    for (int mt = 0; mt < 4; mt++) {
        const _Float16* wrow = wp16 + (size_t)(mt * 16 + ln) * Cc;
        v16h wa0 = mkv(ld16g(wrow + 16 * hi),      ld16g(wrow + 16 * hi + 8));
        v16h wa1 = mkv(ld16g(wrow + 32 + 16 * hi), ld16g(wrow + 40 + 16 * hi));
        v8f z = {};
        v8f d = wmma16(wa0, ob0, wmma16(wa1, ob1, z));
#pragma unroll
        for (int j = 0; j < 8; j++) {
            int oc = mt * 16 + 8 * hi + j;
            size_t idx = (size_t)(b * Cc + oc) * Nn + n0 + ln;
            out[idx] = d[j] + bp[oc] + x[idx];
        }
    }
}

// ---------- launch ----------
extern "C" void kernel_launch(void* const* d_in, const int* in_sizes, int n_in,
                              void* d_out, int out_size, void* d_ws, size_t ws_size,
                              hipStream_t stream) {
    const float* x    = (const float*)d_in[0];
    const float* gn_w = (const float*)d_in[1];
    const float* gn_b = (const float*)d_in[2];
    const float* Wq   = (const float*)d_in[3];
    const float* bq   = (const float*)d_in[4];
    const float* Wk   = (const float*)d_in[5];
    const float* bk   = (const float*)d_in[6];
    const float* Wv   = (const float*)d_in[7];
    const float* bv   = (const float*)d_in[8];
    const float* Wp   = (const float*)d_in[9];
    const float* bp   = (const float*)d_in[10];

    const size_t SZ = (size_t)Bb * Nn * Cc * sizeof(_Float16);   // 4 MiB
    char* ws = (char*)d_ws;
    _Float16* h16  = (_Float16*)(ws + 0 * SZ);
    _Float16* q16  = (_Float16*)(ws + 1 * SZ);
    _Float16* k16  = (_Float16*)(ws + 2 * SZ);
    _Float16* vT16 = (_Float16*)(ws + 3 * SZ);
    _Float16* O16  = (_Float16*)(ws + 4 * SZ);
    _Float16* w16  = (_Float16*)(ws + 5 * SZ);                   // 32 KiB: Wq,Wk,Wv,Wp

    cvtw_kernel<<<16, 256, 0, stream>>>(Wq, Wk, Wv, Wp, w16);
    gn_kernel<<<Bb * Gg, 256, 0, stream>>>(x, gn_w, gn_b, h16);
    qkv_kernel<<<Bb * (Nn / 16), 32, 0, stream>>>(h16, w16, bq, bk, bv, q16, k16, vT16);
    attn_kernel<<<Bb * (Nn / 16), 32, 0, stream>>>(q16, k16, vT16, O16);
    proj_kernel<<<Bb * (Nn / 16), 32, 0, stream>>>(O16, w16 + 3 * 4096, bp, x, (float*)d_out);
}
